// GNN_5214090297538
// MI455X (gfx1250) — compile-verified, matrix operand order checked by reading, not run on hardware
//
#include <hip/hip_runtime.h>
#include <hip/hip_bf16.h>
#include <math.h>

// ---------------------------------------------------------------------------
// Problem constants (match reference)
// ---------------------------------------------------------------------------
#define NN 16384
#define EE 262144
#define HH 256
#define N_LSTM 9

#define ACT_NONE 0
#define ACT_RELU 1
#define ACT_LEAKY 2

typedef __attribute__((ext_vector_type(16))) __bf16 v16bf;
typedef __attribute__((ext_vector_type(8)))  __bf16 bfx8;
typedef __attribute__((ext_vector_type(8)))  float v8f;

// fp32 -> bf16 round-to-nearest-even
__device__ __forceinline__ __bf16 f2bf(float f) {
    unsigned u = __builtin_bit_cast(unsigned, f);
    unsigned r = u + 0x7FFFu + ((u >> 16) & 1u);
    unsigned short h = (unsigned short)(r >> 16);
    return __builtin_bit_cast(__bf16, h);
}

// ---------------------------------------------------------------------------
// Weight/operand packer: D[Rp,Cp](bf16) = S[R,C](f32), zero padded.
// Cp multiple of 32, Rp multiple of 16 -> GEMM needs no K/N guards.
// ---------------------------------------------------------------------------
__global__ void pack_bf16(const float* __restrict__ S, __bf16* __restrict__ D,
                          int R, int C, int Rp, int Cp)
{
    const size_t total = (size_t)Rp * Cp;
    for (size_t i = blockIdx.x * (size_t)blockDim.x + threadIdx.x; i < total;
         i += (size_t)gridDim.x * blockDim.x) {
        const int r = (int)(i / Cp), c = (int)(i % Cp);
        const float v = (r < R && c < C) ? S[(size_t)r * C + c] : 0.f;
        D[i] = f2bf(v);
    }
}

// ---------------------------------------------------------------------------
// WMMA GEMM on pre-packed bf16 operands:
//   C[M,Nc] = act( [A1|A2] @ [B1|B2]^T + bias )
// A1:[M,K1p] A2:[M,K2p]  B1:[Ncp,K1p] B2:[Ncp,K2p]  (bf16, K*p mult of 32,
// rows 16B aligned, zero padded).  One wave = one 16x16 tile.
// Inner loop: 4x global_load_b128 + v_wmma_f32_16x16x32_bf16.
// ---------------------------------------------------------------------------
__global__ void wmma_gemm(const __bf16* __restrict__ A1, const __bf16* __restrict__ A2,
                          const __bf16* __restrict__ B1, const __bf16* __restrict__ B2,
                          const float* __restrict__ bias,
                          float* __restrict__ Cf, __bf16* __restrict__ Cbf,
                          int M, int Nc, int K1p, int K2p, int act)
{
    const int K = K1p + K2p;
    const int tilesN = (Nc + 15) >> 4;
    const int tilesM = M >> 4;
    const int wave = (int)((blockIdx.x * blockDim.x + threadIdx.x) >> 5);
    const int lane = (int)(threadIdx.x & 31);
    const int tm = wave / tilesN;
    const int tn = wave % tilesN;
    if (tm >= tilesM) return;                 // wave-uniform -> EXEC all-ones for WMMA

    const int hf = lane >> 4;
    const int r  = lane & 15;
    const size_t arow = (size_t)(tm * 16 + r);
    const size_t brow = (size_t)(tn * 16 + r);

    v8f acc = {0.f, 0.f, 0.f, 0.f, 0.f, 0.f, 0.f, 0.f};

    for (int kb = 0; kb < K; kb += 32) {
        const __bf16* Ab;
        const __bf16* Bb;
        if (kb < K1p) {
            Ab = A1 + arow * K1p + kb;
            Bb = B1 + brow * K1p + kb;
        } else {
            Ab = A2 + arow * K2p + (kb - K1p);
            Bb = B2 + brow * K2p + (kb - K1p);
        }
        // streamed operand prefetch (global_prefetch_b8)
        __builtin_prefetch((const void*)(Ab + 64), 0, 0);
        const bfx8 al = *(const bfx8*)(Ab + 8 * hf);
        const bfx8 ah = *(const bfx8*)(Ab + 16 + 8 * hf);
        const bfx8 bl = *(const bfx8*)(Bb + 8 * hf);
        const bfx8 bh = *(const bfx8*)(Bb + 16 + 8 * hf);
        const v16bf a = __builtin_shufflevector(al, ah, 0, 1, 2, 3, 4, 5, 6, 7,
                                                8, 9, 10, 11, 12, 13, 14, 15);
        const v16bf b = __builtin_shufflevector(bl, bh, 0, 1, 2, 3, 4, 5, 6, 7,
                                                8, 9, 10, 11, 12, 13, 14, 15);
        acc = __builtin_amdgcn_wmma_f32_16x16x32_bf16(false, a, false, b,
                                                      (short)0, acc, false, false);
    }

    const int n = tn * 16 + (lane & 15);
    if (n < Nc) {
        const float bval = bias ? bias[n] : 0.f;
#pragma unroll
        for (int rr = 0; rr < 8; ++rr) {
            const int m = tm * 16 + rr + (hf << 3);
            float v = acc[rr] + bval;
            if (act == ACT_RELU)       v = v > 0.f ? v : 0.f;
            else if (act == ACT_LEAKY) v = v > 0.f ? v : 0.01f * v;
            if (Cf)  Cf[(size_t)m * Nc + n] = v;
            if (Cbf) Cbf[(size_t)m * Nc + n] = f2bf(v);
        }
    }
}

// ---------------------------------------------------------------------------
// Edge input: X[e,:Dp](bf16) = inorm(concat(nf[src], ef[e], nf[dst])) zero-pad
// one wave per edge (D <= 768 -> 24 regs/lane)
// ---------------------------------------------------------------------------
__global__ void edge_input(const float* __restrict__ nf, int nfd,
                           const float* __restrict__ ef, int efd,
                           const int* __restrict__ src, const int* __restrict__ dst,
                           __bf16* __restrict__ X, int Dp, int E)
{
    const int e = (int)((blockIdx.x * blockDim.x + threadIdx.x) >> 5);
    const int lane = (int)(threadIdx.x & 31);
    if (e >= E) return;
    const int D = 2 * nfd + efd;
    const int s = src[e], d = dst[e];
    float buf[32];
    float sum = 0.f, sq = 0.f;
    for (int idx = lane, t = 0; idx < D; idx += 32, ++t) {
        float v;
        if (idx < nfd)            v = nf[(size_t)s * nfd + idx];
        else if (idx < nfd + efd) v = ef[(size_t)e * efd + (idx - nfd)];
        else                      v = nf[(size_t)d * nfd + (idx - nfd - efd)];
        buf[t] = v; sum += v; sq += v * v;
    }
    for (int m = 16; m > 0; m >>= 1) {
        sum += __shfl_xor(sum, m, 32);
        sq  += __shfl_xor(sq,  m, 32);
    }
    const float mu = sum / (float)D;
    const float var = sq / (float)D - mu * mu;
    const float rs = rsqrtf(var + 1e-5f);
    for (int idx = lane, t = 0; idx < Dp; idx += 32, ++t) {
        const float v = (idx < D) ? (buf[t] - mu) * rs : 0.f;
        X[(size_t)e * Dp + idx] = f2bf(v);
    }
}

// X[m,:Dp](bf16) = inorm(concat(A[m,:d1], B[m,:d2])) zero-pad (wave per row)
__global__ void concat2_inorm(const float* __restrict__ A, int d1,
                              const float* __restrict__ B, int d2,
                              __bf16* __restrict__ X, int Dp, int M)
{
    const int row = (int)((blockIdx.x * blockDim.x + threadIdx.x) >> 5);
    const int lane = (int)(threadIdx.x & 31);
    if (row >= M) return;
    const int D = d1 + d2;
    float buf[32];
    float sum = 0.f, sq = 0.f;
    for (int idx = lane, t = 0; idx < D; idx += 32, ++t) {
        const float v = (idx < d1) ? A[(size_t)row * d1 + idx]
                                   : B[(size_t)row * d2 + (idx - d1)];
        buf[t] = v; sum += v; sq += v * v;
    }
    for (int m = 16; m > 0; m >>= 1) {
        sum += __shfl_xor(sum, m, 32);
        sq  += __shfl_xor(sq,  m, 32);
    }
    const float mu = sum / (float)D;
    const float var = sq / (float)D - mu * mu;
    const float rs = rsqrtf(var + 1e-5f);
    for (int idx = lane, t = 0; idx < Dp; idx += 32, ++t) {
        const float v = (idx < D) ? (buf[t] - mu) * rs : 0.f;
        X[(size_t)row * Dp + idx] = f2bf(v);
    }
}

// Y[m,:D](bf16) = inorm(X[m,:D])   (wave per row, D mult of 32)
__global__ void inorm_rows_bf(const float* __restrict__ X, __bf16* __restrict__ Y,
                              int M, int D)
{
    const int row = (int)((blockIdx.x * blockDim.x + threadIdx.x) >> 5);
    const int lane = (int)(threadIdx.x & 31);
    if (row >= M) return;
    float buf[32];
    float sum = 0.f, sq = 0.f;
    for (int idx = lane, t = 0; idx < D; idx += 32, ++t) {
        const float v = X[(size_t)row * D + idx];
        buf[t] = v; sum += v; sq += v * v;
    }
    for (int m = 16; m > 0; m >>= 1) {
        sum += __shfl_xor(sum, m, 32);
        sq  += __shfl_xor(sq,  m, 32);
    }
    const float mu = sum / (float)D;
    const float var = sq / (float)D - mu * mu;
    const float rs = rsqrtf(var + 1e-5f);
    for (int idx = lane, t = 0; idx < D; idx += 32, ++t)
        Y[(size_t)row * D + idx] = f2bf((buf[t] - mu) * rs);
}

// atom pre-stage: xb[n,:32](bf16) = mean_a inorm(atom_emb[n,a,:7]) zero-pad
__global__ void atom_xbar(const float* __restrict__ ae, __bf16* __restrict__ xb, int n)
{
    const int i = (int)(blockIdx.x * blockDim.x + threadIdx.x);
    if (i >= n) return;
    float acc[7] = {0.f, 0.f, 0.f, 0.f, 0.f, 0.f, 0.f};
    for (int a = 0; a < 14; ++a) {
        float v[7]; float s = 0.f;
        for (int j = 0; j < 7; ++j) { v[j] = ae[((size_t)i * 14 + a) * 7 + j]; s += v[j]; }
        const float mu = s * (1.f / 7.f);
        float q = 0.f;
        for (int j = 0; j < 7; ++j) { const float d = v[j] - mu; q += d * d; }
        const float rs = rsqrtf(q * (1.f / 7.f) + 1e-5f);
        for (int j = 0; j < 7; ++j) acc[j] += (v[j] - mu) * rs;
    }
    for (int j = 0; j < 7; ++j)  xb[(size_t)i * 32 + j] = f2bf(acc[j] * (1.f / 14.f));
    for (int j = 7; j < 32; ++j) xb[(size_t)i * 32 + j] = f2bf(0.f);
}

// X[m,:] = concat(A[m,:d1], B[m,:d2])  (fp32, element-per-thread)
__global__ void concat_rows(const float* __restrict__ A, int d1,
                            const float* __restrict__ B, int d2,
                            float* __restrict__ X, int M)
{
    const int D = d1 + d2;
    const size_t total = (size_t)M * D;
    for (size_t i = blockIdx.x * (size_t)blockDim.x + threadIdx.x; i < total;
         i += (size_t)gridDim.x * blockDim.x) {
        const int m = (int)(i / D), j = (int)(i % D);
        X[i] = (j < d1) ? A[(size_t)m * d1 + j] : B[(size_t)m * d2 + (j - d1)];
    }
}

// LSTM cell: gates G[m,4H] (i,f,g,o); in-place h(fp32)+c; bf16 mirror of h
__global__ void lstm_apply(const float* __restrict__ G, float* __restrict__ h,
                           __bf16* __restrict__ hbf, float* __restrict__ c, int M)
{
    const size_t idx = blockIdx.x * (size_t)blockDim.x + threadIdx.x;
    if (idx >= (size_t)M * HH) return;
    const size_t m = idx >> 8; const int j = (int)(idx & 255);
    const float* g = G + m * (4 * HH);
    const float gi = g[j], gf = g[HH + j], gg = g[2 * HH + j], go = g[3 * HH + j];
    const float si = 1.f / (1.f + expf(-gi));
    const float sf = 1.f / (1.f + expf(-gf));
    const float so = 1.f / (1.f + expf(-go));
    const float c2 = sf * c[idx] + si * tanhf(gg);
    c[idx] = c2;
    const float hv = so * tanhf(c2);
    h[idx] = hv;
    hbf[idx] = f2bf(hv);
}

// attn[m] = dot(X[m,:D], w[:D])  (wave per row)
__global__ void rowdot(const float* __restrict__ X, const float* __restrict__ w,
                       float* __restrict__ out, int M, int D)
{
    const int row = (int)((blockIdx.x * blockDim.x + threadIdx.x) >> 5);
    const int lane = (int)(threadIdx.x & 31);
    if (row >= M) return;
    float s = 0.f;
    for (int j = lane; j < D; j += 32) s += X[(size_t)row * D + j] * w[j];
    for (int m = 16; m > 0; m >>= 1) s += __shfl_xor(s, m, 32);
    if (lane == 0) out[row] = s;
}

// ----- segment softmax helpers (monotonic uint keys for float atomicMax) -----
__device__ __forceinline__ unsigned fkey(float x) {
    unsigned u = __builtin_bit_cast(unsigned, x);
    return (u & 0x80000000u) ? ~u : (u | 0x80000000u);
}
__device__ __forceinline__ float funkey(unsigned k) {
    unsigned u = (k & 0x80000000u) ? (k & 0x7FFFFFFFu) : ~k;
    return __builtin_bit_cast(float, u);
}

__global__ void seg_max(const float* __restrict__ attn, const int* __restrict__ dst,
                        unsigned* __restrict__ mkey, int E)
{
    const int e = (int)(blockIdx.x * blockDim.x + threadIdx.x);
    if (e >= E) return;
    atomicMax(&mkey[dst[e]], fkey(attn[e]));
}

__global__ void seg_m_final(const unsigned* __restrict__ mkey, float* __restrict__ m, int n)
{
    const int i = (int)(blockIdx.x * blockDim.x + threadIdx.x);
    if (i >= n) return;
    const unsigned k = mkey[i];
    m[i] = (k == 0u) ? 0.f : funkey(k);   // 0-degree guard (matches reference)
}

__global__ void seg_accum(const float* __restrict__ attn, const float* __restrict__ feat,
                          const int* __restrict__ dst, const float* __restrict__ m,
                          float* __restrict__ den, float* __restrict__ agg, int E)
{
    const int e = (int)((blockIdx.x * blockDim.x + threadIdx.x) >> 5);
    const int lane = (int)(threadIdx.x & 31);
    if (e >= E) return;
    const int d = dst[e];
    const float ev = expf(attn[e] - m[d]);
    if (lane == 0) atomicAdd(&den[d], ev);
    for (int j = lane; j < HH; j += 32)
        atomicAdd(&agg[(size_t)d * HH + j], ev * feat[(size_t)e * HH + j]);
}

__global__ void seg_norm(float* __restrict__ agg, const float* __restrict__ den, int n)
{
    const size_t idx = blockIdx.x * (size_t)blockDim.x + threadIdx.x;
    if (idx >= (size_t)n * HH) return;
    const float dd = den[idx >> 8];
    if (dd > 0.f) agg[idx] /= dd;
}

// ---------------------------------------------------------------------------
// host-side orchestration
// ---------------------------------------------------------------------------
static inline int gemm_blocks(int M, int Nc) {
    long long tiles = (long long)(M >> 4) * ((Nc + 15) >> 4);
    return (int)((tiles + 7) / 8);            // 8 waves (256 threads) per block
}
static inline int row_blocks(int rows) { return (rows + 7) / 8; }

extern "C" void kernel_launch(void* const* d_in, const int* in_sizes, int n_in,
                              void* d_out, int out_size, void* d_ws, size_t ws_size,
                              hipStream_t stream)
{
    (void)in_sizes; (void)n_in; (void)out_size; (void)ws_size;
    const float* nfeat   = (const float*)d_in[0];   // [N,28]
    const float* atomemb = (const float*)d_in[1];   // [N,14,7]
    const float* efeat0  = (const float*)d_in[2];   // [E,15]
    const int*   src     = (const int*)d_in[3];
    const int*   dst     = (const int*)d_in[4];
    const float* atom_W1 = (const float*)d_in[5];   // [256,7]
    const float* atom_W2 = (const float*)d_in[6];   // [256,256]
    const float* e1_W    = (const float*)d_in[7];   // [256,583]
    const float* e1_b    = (const float*)d_in[8];
    const float* e1_attn = (const float*)d_in[9];   // [1,256]
    const float* n1_W    = (const float*)d_in[10];  // [256,540]
    const float* n1_b    = (const float*)d_in[11];
    const float* eW      = (const float*)d_in[12];  // [9,256,768]
    const float* eb      = (const float*)d_in[13];  // [9,256]
    const float* eWih    = (const float*)d_in[14];  // [9,1024,256]
    const float* eWhh    = (const float*)d_in[15];  // [9,1024,256]
    const float* eattn   = (const float*)d_in[16];  // [9,1,256]
    const float* nW      = (const float*)d_in[17];  // [9,256,512]
    const float* nb      = (const float*)d_in[18];  // [9,256]
    const float* nWih    = (const float*)d_in[19];  // [9,1024,256]
    const float* nWhh    = (const float*)d_in[20];  // [9,1024,256]
    const float* out_W   = (const float*)d_in[21];  // [37,256]
    const float* out_b   = (const float*)d_in[22];  // [37]
    float* out = (float*)d_out;                     // [E,37]

    // ---- workspace carve-up ----
    float* ws = (float*)d_ws;
    size_t off = 0;
    auto allocf = [&](size_t n) { float* p = ws + off; off += (n + 7) & ~(size_t)7; return p; };
    auto allocb = [&](size_t n) { __bf16* p = (__bf16*)(ws + off);
                                  off += (((n + 1) / 2) + 7) & ~(size_t)7; return p; };
    // fp32 state
    float* nfA  = allocf((size_t)NN * 284);   // layer-1 node features
    float* nfB  = allocf((size_t)NN * HH);    // node h
    float* nfC  = allocf((size_t)NN * HH);    // node cell
    float* agg  = allocf((size_t)NN * HH);
    float* a1   = allocf((size_t)NN * HH);
    float* a2   = allocf((size_t)NN * HH);
    float* nG   = allocf((size_t)NN * 1024);
    float* ebufA = allocf((size_t)EE * HH);   // edge h
    float* efC  = allocf((size_t)EE * HH);    // edge cell
    float* G    = allocf((size_t)EE * 1024);
    float* mval = allocf(NN);
    float* den  = allocf(NN);
    float* attn = allocf(EE);
    unsigned* mkey = (unsigned*)allocf(NN);
    // bf16 activations (GEMM operands)
    __bf16* Xe_bf   = allocb((size_t)EE * 768);   // edge GEMM input (stride 608/768)
    __bf16* nXin_bf = allocb((size_t)NN * 544);   // node GEMM input (stride 544/512)
    __bf16* ex_bf   = allocb((size_t)EE * HH);    // edge pre-LSTM x
    __bf16* ef_bf   = allocb((size_t)EE * HH);    // edge h mirror
    __bf16* xn_bf   = allocb((size_t)NN * HH);    // node pre-LSTM x
    __bf16* nf_bf   = allocb((size_t)NN * HH);    // node h mirror
    __bf16* a1n_bf  = allocb((size_t)NN * HH);
    __bf16* xbar_bf = allocb((size_t)NN * 32);
    // bf16 packed weights
    __bf16* pW_a1  = allocb((size_t)256 * 32);
    __bf16* pW_a2  = allocb((size_t)256 * 256);
    __bf16* pW_e1  = allocb((size_t)256 * 608);
    __bf16* pW_n1  = allocb((size_t)256 * 544);
    __bf16* pW_eW  = allocb((size_t)9 * 256 * 768);
    __bf16* pW_eWih = allocb((size_t)9 * 1024 * 256);
    __bf16* pW_eWhh = allocb((size_t)9 * 1024 * 256);
    __bf16* pW_nW  = allocb((size_t)9 * 256 * 512);
    __bf16* pW_nWih = allocb((size_t)9 * 1024 * 256);
    __bf16* pW_nWhh = allocb((size_t)9 * 1024 * 256);
    __bf16* pW_out = allocb((size_t)48 * 256);

    const int T = 256;

    // ---- pack weights to bf16 (padded) ----
    pack_bf16<<<512, T, 0, stream>>>(atom_W1, pW_a1, 256, 7, 256, 32);
    pack_bf16<<<512, T, 0, stream>>>(atom_W2, pW_a2, 256, 256, 256, 256);
    pack_bf16<<<1024, T, 0, stream>>>(e1_W, pW_e1, 256, 583, 256, 608);
    pack_bf16<<<1024, T, 0, stream>>>(n1_W, pW_n1, 256, 540, 256, 544);
    pack_bf16<<<2048, T, 0, stream>>>(eW, pW_eW, 9 * 256, 768, 9 * 256, 768);
    pack_bf16<<<2048, T, 0, stream>>>(eWih, pW_eWih, 9 * 1024, 256, 9 * 1024, 256);
    pack_bf16<<<2048, T, 0, stream>>>(eWhh, pW_eWhh, 9 * 1024, 256, 9 * 1024, 256);
    pack_bf16<<<2048, T, 0, stream>>>(nW, pW_nW, 9 * 256, 512, 9 * 256, 512);
    pack_bf16<<<2048, T, 0, stream>>>(nWih, pW_nWih, 9 * 1024, 256, 9 * 1024, 256);
    pack_bf16<<<2048, T, 0, stream>>>(nWhh, pW_nWhh, 9 * 1024, 256, 9 * 1024, 256);
    pack_bf16<<<64, T, 0, stream>>>(out_W, pW_out, 37, 256, 48, 256);

    // ---- atom embedding ----
    atom_xbar<<<(NN + T - 1) / T, T, 0, stream>>>(atomemb, xbar_bf, NN);
    wmma_gemm<<<gemm_blocks(NN, HH), T, 0, stream>>>(xbar_bf, nullptr, pW_a1, nullptr,
                                                     nullptr, a1, nullptr,
                                                     NN, HH, 32, 0, ACT_RELU);
    inorm_rows_bf<<<row_blocks(NN), T, 0, stream>>>(a1, a1n_bf, NN, HH);
    wmma_gemm<<<gemm_blocks(NN, HH), T, 0, stream>>>(a1n_bf, nullptr, pW_a2, nullptr,
                                                     nullptr, a2, nullptr,
                                                     NN, HH, 256, 0, ACT_RELU);
    concat_rows<<<4096, T, 0, stream>>>(nfeat, 28, a2, HH, nfA, NN);   // nfA [N,284]

    // ---- zero cell states ----
    hipMemsetAsync(efC, 0, (size_t)EE * HH * sizeof(float), stream);
    hipMemsetAsync(nfC, 0, (size_t)NN * HH * sizeof(float), stream);

    auto seg_softmax = [&](const float* feat) {
        hipMemsetAsync(mkey, 0, (size_t)NN * sizeof(unsigned), stream);
        hipMemsetAsync(den, 0, (size_t)NN * sizeof(float), stream);
        hipMemsetAsync(agg, 0, (size_t)NN * HH * sizeof(float), stream);
        seg_max<<<(EE + T - 1) / T, T, 0, stream>>>(attn, dst, mkey, EE);
        seg_m_final<<<(NN + T - 1) / T, T, 0, stream>>>(mkey, mval, NN);
        seg_accum<<<row_blocks(EE), T, 0, stream>>>(attn, feat, dst, mval, den, agg, EE);
        seg_norm<<<((size_t)NN * HH + T - 1) / T, T, 0, stream>>>(agg, den, NN);
    };

    // ---- layer 1 (no LSTM): h = leaky(inorm(concat)@e1_W^T+b) ----
    edge_input<<<row_blocks(EE), T, 0, stream>>>(nfA, 284, efeat0, 15, src, dst,
                                                 Xe_bf, 608, EE);
    wmma_gemm<<<gemm_blocks(EE, HH), T, 0, stream>>>(Xe_bf, nullptr, pW_e1, nullptr,
                                                     e1_b, ebufA, ef_bf,
                                                     EE, HH, 608, 0, ACT_LEAKY);
    rowdot<<<row_blocks(EE), T, 0, stream>>>(ebufA, e1_attn, attn, EE, HH);
    seg_softmax(ebufA);
    concat2_inorm<<<row_blocks(NN), T, 0, stream>>>(nfA, 284, agg, HH, nXin_bf, 544, NN);
    wmma_gemm<<<gemm_blocks(NN, HH), T, 0, stream>>>(nXin_bf, nullptr, pW_n1, nullptr,
                                                     n1_b, nfB, nf_bf,
                                                     NN, HH, 544, 0, ACT_RELU);

    // ---- 9 LSTM message-passing layers ----
    for (int i = 0; i < N_LSTM; ++i) {
        const __bf16* eWi   = pW_eW   + (size_t)i * 256 * 768;
        const float*  ebi   = eb      + (size_t)i * HH;
        const __bf16* eWihi = pW_eWih + (size_t)i * 1024 * 256;
        const __bf16* eWhhi = pW_eWhh + (size_t)i * 1024 * 256;
        const float*  eati  = eattn   + (size_t)i * HH;
        const __bf16* nWi   = pW_nW   + (size_t)i * 256 * 512;
        const float*  nbi   = nb      + (size_t)i * HH;
        const __bf16* nWihi = pW_nWih + (size_t)i * 1024 * 256;
        const __bf16* nWhhi = pW_nWhh + (size_t)i * 1024 * 256;

        // edge update + edge LSTM
        edge_input<<<row_blocks(EE), T, 0, stream>>>(nfB, HH, ebufA, HH, src, dst,
                                                     Xe_bf, 768, EE);
        wmma_gemm<<<gemm_blocks(EE, HH), T, 0, stream>>>(Xe_bf, nullptr, eWi, nullptr,
                                                         ebi, nullptr, ex_bf,
                                                         EE, HH, 768, 0, ACT_LEAKY);
        wmma_gemm<<<gemm_blocks(EE, 1024), T, 0, stream>>>(ex_bf, ef_bf, eWihi, eWhhi,
                                                           nullptr, G, nullptr,
                                                           EE, 1024, 256, 256, ACT_NONE);
        lstm_apply<<<((size_t)EE * HH + T - 1) / T, T, 0, stream>>>(G, ebufA, ef_bf,
                                                                    efC, EE);
        // attention aggregation
        rowdot<<<row_blocks(EE), T, 0, stream>>>(ebufA, eati, attn, EE, HH);
        seg_softmax(ebufA);

        // node update + node LSTM
        concat2_inorm<<<row_blocks(NN), T, 0, stream>>>(nfB, HH, agg, HH,
                                                        nXin_bf, 512, NN);
        wmma_gemm<<<gemm_blocks(NN, HH), T, 0, stream>>>(nXin_bf, nullptr, nWi, nullptr,
                                                         nbi, nullptr, xn_bf,
                                                         NN, HH, 512, 0, ACT_RELU);
        wmma_gemm<<<gemm_blocks(NN, 1024), T, 0, stream>>>(xn_bf, nf_bf, nWihi, nWhhi,
                                                           nullptr, nG, nullptr,
                                                           NN, 1024, 256, 256, ACT_NONE);
        lstm_apply<<<((size_t)NN * HH + T - 1) / T, T, 0, stream>>>(nG, nfB, nf_bf,
                                                                    nfC, NN);
    }

    // ---- final extra edge update (layer-8 params), no aggregation ----
    {
        const int i = N_LSTM - 1;
        const __bf16* eWi   = pW_eW   + (size_t)i * 256 * 768;
        const float*  ebi   = eb      + (size_t)i * HH;
        const __bf16* eWihi = pW_eWih + (size_t)i * 1024 * 256;
        const __bf16* eWhhi = pW_eWhh + (size_t)i * 1024 * 256;
        edge_input<<<row_blocks(EE), T, 0, stream>>>(nfB, HH, ebufA, HH, src, dst,
                                                     Xe_bf, 768, EE);
        wmma_gemm<<<gemm_blocks(EE, HH), T, 0, stream>>>(Xe_bf, nullptr, eWi, nullptr,
                                                         ebi, nullptr, ex_bf,
                                                         EE, HH, 768, 0, ACT_LEAKY);
        wmma_gemm<<<gemm_blocks(EE, 1024), T, 0, stream>>>(ex_bf, ef_bf, eWihi, eWhhi,
                                                           nullptr, G, nullptr,
                                                           EE, 1024, 256, 256, ACT_NONE);
        lstm_apply<<<((size_t)EE * HH + T - 1) / T, T, 0, stream>>>(G, ebufA, ef_bf,
                                                                    efC, EE);
    }

    // ---- output projection: out[E,37] = h @ out_W^T + out_b ----
    wmma_gemm<<<gemm_blocks(EE, 37), T, 0, stream>>>(ef_bf, nullptr, pW_out, nullptr,
                                                     out_b, out, nullptr,
                                                     EE, 37, 256, 0, ACT_NONE);
}
